// MultiheadAttentionWithRPE_41592463294728
// MI455X (gfx1250) — compile-verified, matrix operand order checked by reading, not run on hardware
//
#include <hip/hip_runtime.h>

typedef __attribute__((ext_vector_type(16))) __bf16 v16bf;
typedef __attribute__((ext_vector_type(8)))  float  v8f;
typedef unsigned int       u32;
typedef unsigned long long u64;
typedef u32 u32x4 __attribute__((ext_vector_type(4)));
typedef int i32x4 __attribute__((ext_vector_type(4)));
typedef int i32x8 __attribute__((ext_vector_type(8)));

#define BATCH 4
#define SLEN 1024
#define EMB 1024
#define NH 16
#define HD 64
#define MROWS (BATCH * SLEN)   // 4096
#define TPAD 2048              // padded table rows per head (need 0..2047)

__device__ __forceinline__ v16bf load16(const __bf16* p) {
  // A/B fragment per-lane pattern: 8 contiguous bf16 at p, 8 more at p+16.
  union { uint4 u[2]; v16bf v; } r;
  r.u[0] = *(const uint4*)(p);
  r.u[1] = *(const uint4*)(p + 16);
  return r.v;
}

__device__ __forceinline__ v8f wmma_bf16(v16bf a, v16bf b, v8f c) {
  return __builtin_amdgcn_wmma_f32_16x16x32_bf16(
      /*neg_a=*/false, a, /*neg_b=*/false, b,
      /*c_mod=*/(short)0, c, /*reuse_a=*/false, /*reuse_b=*/false);
}

// Tensor Data Mover: 2-D tile (tile_d0 contiguous elems x tile_d1 rows) of
// bf16 from global (row stride stride_d0 elems) into contiguous LDS.
// D# bitfields per CDNA5 ISA 8.3/8.4 (group0: count/lds/global/type,
// group1: data_size, tensor dims, tile dims, dim0 stride).
__device__ __forceinline__ void tdm_load_2d(u32 lds_off, const void* gaddr,
                                            u32 tensor_d0, u32 tensor_d1,
                                            u32 stride_d0, u32 tile_d0,
                                            u32 tile_d1) {
  u64 ga = (u64)(uintptr_t)gaddr;
  u32x4 g0;
  g0[0] = 1u;                                            // count=1
  g0[1] = lds_off;                                       // lds_addr
  g0[2] = (u32)ga;                                       // global_addr[31:0]
  g0[3] = (u32)((ga >> 32) & 0x1FFFFFFu) | (2u << 30);   // addr[56:32] | type=2
  i32x8 g1;
  g1[0] = (int)(1u << 16);                               // data_size=1 (2B)
  g1[1] = (int)((tensor_d0 & 0xFFFFu) << 16);            // tensor_dim0[15:0]
  g1[2] = (int)((tensor_d0 >> 16) | ((tensor_d1 & 0xFFFFu) << 16));
  g1[3] = (int)((tensor_d1 >> 16) | (tile_d0 << 16));    // tile_dim0
  g1[4] = (int)(tile_d1 & 0xFFFFu);                      // tile_dim1, tile_dim2=0
  g1[5] = (int)stride_d0;                                // tensor_dim0_stride[31:0]
  g1[6] = 0;
  g1[7] = 0;
  i32x4 z4 = {};
#if __clang_major__ >= 23
  i32x8 z8 = {};
  __builtin_amdgcn_tensor_load_to_lds(g0, g1, z4, z4, z8, 0);
#else
  __builtin_amdgcn_tensor_load_to_lds(g0, g1, z4, z4, 0);
#endif
}

// ---------------------------------------------------------------- casts
__global__ void cast_f32_bf16(const float* __restrict__ src,
                              __bf16* __restrict__ dst, int n) {
  int i = blockIdx.x * blockDim.x + threadIdx.x;
  if (i < n) dst[i] = (__bf16)src[i];
}

// src: H x 9999 x 64, dst: H x 2048 x 64 (rows >= 2047 zero padded)
__global__ void cast_table_bf16(const float* __restrict__ src,
                                __bf16* __restrict__ dst) {
  int i = blockIdx.x * blockDim.x + threadIdx.x;  // NH*TPAD*HD elements
  int d = i & 63;
  int r = (i >> 6) & (TPAD - 1);
  int h = i >> 17;
  float v = (r < 2 * SLEN - 1) ? src[((size_t)h * 9999 + r) * HD + d] : 0.0f;
  dst[i] = (__bf16)v;
}

// ---------------------------------------------------------------- GEMM
// Y[m][n] = sum_k A[m][k] * Wt[n][k] + bias[n]
// mode 0: store bf16 into (B,H,S,HD) layout (QKV projections)
// mode 1: store f32 row-major M x N (output projection)
#define TM 128
#define TN 128
#define KC 32

__global__ __launch_bounds__(256)
void gemm_bf16(const __bf16* __restrict__ A, const __bf16* __restrict__ Wt,
               const float* __restrict__ bias, void* __restrict__ out,
               int K, int N, int mode) {
  __shared__ __bf16 As[TM * KC];   // row-major 128 x 32
  __shared__ __bf16 Bs[TN * KC];   // col-major: 128 cols x 32 k contiguous

  const int tid  = threadIdx.x;
  const int lane = tid & 31;
  const int wave = tid >> 5;
  const int m0 = blockIdx.y * TM;
  const int n0 = blockIdx.x * TN;
  const int wm = wave & 3;   // 4 waves along M (32 rows each)
  const int wn = wave >> 2;  // 2 waves along N (64 cols each)
  const int jn    = lane & 15;
  const int half8 = ((lane >> 4) & 1) * 8;

  v8f acc[2][4];
#pragma unroll
  for (int ms = 0; ms < 2; ms++)
#pragma unroll
    for (int ns = 0; ns < 4; ns++) acc[ms][ns] = v8f{};

  for (int kk = 0; kk < K; kk += KC) {
    __syncthreads();
    if (wave == 0) {
      // Tensor Data Mover: stage both tiles (8 KB each) without touching VGPRs
      tdm_load_2d((u32)(uintptr_t)&As[0], &A[(size_t)m0 * K + kk],
                  (u32)K, (u32)MROWS, (u32)K, KC, TM);
      tdm_load_2d((u32)(uintptr_t)&Bs[0], &Wt[(size_t)n0 * K + kk],
                  (u32)K, (u32)EMB, (u32)K, KC, TN);
      __builtin_amdgcn_s_wait_tensorcnt(0);
    }
    __syncthreads();

    v16bf af[2], bfr[4];
#pragma unroll
    for (int ms = 0; ms < 2; ms++)
      af[ms] = load16(&As[(wm * 32 + ms * 16 + jn) * KC + half8]);
#pragma unroll
    for (int ns = 0; ns < 4; ns++)
      bfr[ns] = load16(&Bs[(wn * 64 + ns * 16 + jn) * KC + half8]);
#pragma unroll
    for (int ms = 0; ms < 2; ms++)
#pragma unroll
      for (int ns = 0; ns < 4; ns++)
        acc[ms][ns] = wmma_bf16(af[ms], bfr[ns], acc[ms][ns]);
  }

#pragma unroll
  for (int ns = 0; ns < 4; ns++) {
    float bv = bias[n0 + wn * 64 + ns * 16 + jn];
#pragma unroll
    for (int ms = 0; ms < 2; ms++)
#pragma unroll
      for (int v = 0; v < 8; v++) acc[ms][ns][v] += bv;
  }

#pragma unroll
  for (int ms = 0; ms < 2; ms++)
#pragma unroll
    for (int ns = 0; ns < 4; ns++)
#pragma unroll
      for (int v = 0; v < 8; v++) {
        int m = m0 + wm * 32 + ms * 16 + v + half8;
        int n = n0 + wn * 64 + ns * 16 + jn;
        float val = acc[ms][ns][v];
        if (mode == 0) {
          int b = m >> 10, s = m & (SLEN - 1);
          int h = n >> 6,  d = n & 63;
          ((__bf16*)out)[(((size_t)b * NH + h) * SLEN + s) * HD + d] = (__bf16)val;
        } else {
          ((float*)out)[(size_t)m * N + n] = val;
        }
      }
}

// ------------------------------------------------- fused flash attn + RPE
// grid: (B*H, S/64), block: 128 (4 waves, 16 q-rows each)
__global__ __launch_bounds__(128)
void attn_rpe(const __bf16* __restrict__ qg, const __bf16* __restrict__ kg,
              const __bf16* __restrict__ vg, const __bf16* __restrict__ table,
              __bf16* __restrict__ ctx, float scale) {
  __shared__ __bf16 Vt[HD * 32];       // col-major: [d][k], stride 32
  __shared__ __bf16 Ps[4][16 * 32];    // per-wave P staging, row-major stride 32

  const int tid  = threadIdx.x;
  const int lane = tid & 31;
  const int wave = tid >> 5;
  const int bh = blockIdx.x;
  const int b  = bh >> 4;
  const int h  = bh & 15;
  const int q0 = blockIdx.y * 64 + wave * 16;
  const int jn    = lane & 15;
  const int half8 = ((lane >> 4) & 1) * 8;

  const __bf16* kbase = kg + (size_t)bh * SLEN * HD;
  const __bf16* vbase = vg + (size_t)bh * SLEN * HD;
  const __bf16* tbase = table + (size_t)h * TPAD * HD;

  // Q fragments for this wave's 16 rows (A layout), loaded once
  const __bf16* qp = qg + ((size_t)bh * SLEN + q0 + jn) * HD;
  v16bf qa0 = load16(qp + half8);
  v16bf qa1 = load16(qp + 32 + half8);

  float mrun[8], lrun[8];
#pragma unroll
  for (int v = 0; v < 8; v++) { mrun[v] = -1e30f; lrun[v] = 0.0f; }
  v8f o[4];
#pragma unroll
  for (int ns = 0; ns < 4; ns++) o[ns] = v8f{};

  for (int k0 = 0; k0 < SLEN; k0 += 32) {
    __syncthreads();
    // cooperative V tile load (32 k-rows x 64 d), transposed into LDS
#pragma unroll
    for (int i = 0; i < 2; i++) {
      int idx = tid + i * 128;       // 256 uint4 total
      int kr  = idx >> 3;            // 0..31
      int c8  = (idx & 7) * 8;       // 0..56
      union { uint4 u; __bf16 e[8]; } t;
      t.u = *(const uint4*)&vbase[(size_t)(k0 + kr) * HD + c8];
#pragma unroll
      for (int d8 = 0; d8 < 8; d8++) Vt[(c8 + d8) * 32 + kr] = t.e[d8];
    }
    __syncthreads();

    if (k0 + 32 < SLEN) {  // prefetch next K tile (global_prefetch_b8)
      __builtin_prefetch(kbase + (size_t)(k0 + 32 + lane) * HD, 0, 1);
    }

    float sc[2][8];
#pragma unroll
    for (int c = 0; c < 2; c++) {
      // scores subtile: Q (16xHD) @ K^T, cols k0+16c .. +15
      const __bf16* kp = kbase + (size_t)(k0 + c * 16 + jn) * HD;
      v8f s = v8f{};
      s = wmma_bf16(qa0, load16(kp + half8), s);
      s = wmma_bf16(qa1, load16(kp + 32 + half8), s);

      // RPE rel tiles: table rows rbase .. rbase+31
      int rbase = q0 - (k0 + c * 16) + (SLEN - 1) - 15;
      const __bf16* t0 = tbase + (size_t)(rbase + jn) * HD;
      const __bf16* t1 = tbase + (size_t)(rbase + 16 + jn) * HD;
      v8f r0 = v8f{}, r1 = v8f{};
      r0 = wmma_bf16(qa0, load16(t0 + half8), r0);
      r0 = wmma_bf16(qa1, load16(t0 + 32 + half8), r0);
      r1 = wmma_bf16(qa0, load16(t1 + half8), r1);
      r1 = wmma_bf16(qa1, load16(t1 + 32 + half8), r1);

      // gather diagonal rel element: (i,j) needs t = i-j+15 in [0,30]
#pragma unroll
      for (int v = 0; v < 8; v++) {
        int i = v + half8;
        int t = i - jn + 15;
        int src = (t & 15) | (lane & 16);
        float va = __shfl(r0[v], src, 32);
        float vb = __shfl(r1[v], src, 32);
        sc[c][v] = s[v] * scale + ((t < 16) ? va : vb);
      }
    }

    // online softmax over the 32 columns
    float alpha[8];
#pragma unroll
    for (int v = 0; v < 8; v++) {
      float mloc = fmaxf(sc[0][v], sc[1][v]);
#pragma unroll
      for (int off = 8; off >= 1; off >>= 1)
        mloc = fmaxf(mloc, __shfl_xor(mloc, off, 16));
      float mnew = fmaxf(mrun[v], mloc);
      alpha[v] = __expf(mrun[v] - mnew);
      mrun[v] = mnew;
      float p0 = __expf(sc[0][v] - mnew);
      float p1 = __expf(sc[1][v] - mnew);
      sc[0][v] = p0; sc[1][v] = p1;
      float ls = p0 + p1;
#pragma unroll
      for (int off = 8; off >= 1; off >>= 1) ls += __shfl_xor(ls, off, 16);
      lrun[v] = lrun[v] * alpha[v] + ls;
    }
#pragma unroll
    for (int ns = 0; ns < 4; ns++)
#pragma unroll
      for (int v = 0; v < 8; v++) o[ns][v] *= alpha[v];

    // P (C layout) -> LDS -> A-layout fragment
    __bf16* ps = &Ps[wave][0];
#pragma unroll
    for (int c = 0; c < 2; c++)
#pragma unroll
      for (int v = 0; v < 8; v++)
        ps[(v + half8) * 32 + c * 16 + jn] = (__bf16)sc[c][v];
    v16bf pf = load16(ps + jn * 32 + half8);

    // O += P (16x32) @ V (32x64)
#pragma unroll
    for (int ns = 0; ns < 4; ns++) {
      v16bf vf = load16(&Vt[(ns * 16 + jn) * 32 + half8]);
      o[ns] = wmma_bf16(pf, vf, o[ns]);
    }
  }

  // normalize and write ctx in (B,S,E) layout (bf16) for output projection
#pragma unroll
  for (int v = 0; v < 8; v++) {
    float inv = 1.0f / lrun[v];
#pragma unroll
    for (int ns = 0; ns < 4; ns++) {
      int i = v + half8;
      ctx[((size_t)(b * SLEN + q0 + i)) * EMB + h * HD + ns * 16 + jn] =
          (__bf16)(o[ns][v] * inv);
    }
  }
}

// ---------------------------------------------------------------- launch
extern "C" void kernel_launch(void* const* d_in, const int* in_sizes, int n_in,
                              void* d_out, int out_size, void* d_ws,
                              size_t ws_size, hipStream_t stream) {
  const float* x   = (const float*)d_in[0];
  const float* wq  = (const float*)d_in[1];
  const float* bq  = (const float*)d_in[2];
  const float* wk  = (const float*)d_in[3];
  const float* bk  = (const float*)d_in[4];
  const float* wv  = (const float*)d_in[5];
  const float* bv  = (const float*)d_in[6];
  const float* wo  = (const float*)d_in[7];
  const float* bo  = (const float*)d_in[8];
  const float* rpe = (const float*)d_in[9];

  char* ws = (char*)d_ws;
  size_t off = 0;
  auto alloc = [&](size_t bytes) -> void* {
    void* p = ws + off;
    off += (bytes + 255) & ~(size_t)255;
    return p;
  };
  __bf16* xb  = (__bf16*)alloc((size_t)MROWS * EMB * 2);
  __bf16* wqb = (__bf16*)alloc((size_t)EMB * EMB * 2);
  __bf16* wkb = (__bf16*)alloc((size_t)EMB * EMB * 2);
  __bf16* wvb = (__bf16*)alloc((size_t)EMB * EMB * 2);
  __bf16* wob = (__bf16*)alloc((size_t)EMB * EMB * 2);
  __bf16* tb  = (__bf16*)alloc((size_t)NH * TPAD * HD * 2);
  __bf16* qb  = (__bf16*)alloc((size_t)MROWS * EMB * 2);
  __bf16* kb  = (__bf16*)alloc((size_t)MROWS * EMB * 2);
  __bf16* vb  = (__bf16*)alloc((size_t)MROWS * EMB * 2);
  __bf16* cb  = (__bf16*)alloc((size_t)MROWS * EMB * 2);

  const int nX = MROWS * EMB;  // 4194304
  const int nW = EMB * EMB;    // 1048576
  const int nT = NH * TPAD * HD;

  cast_f32_bf16<<<(nX + 255) / 256, 256, 0, stream>>>(x, xb, nX);
  cast_f32_bf16<<<(nW + 255) / 256, 256, 0, stream>>>(wq, wqb, nW);
  cast_f32_bf16<<<(nW + 255) / 256, 256, 0, stream>>>(wk, wkb, nW);
  cast_f32_bf16<<<(nW + 255) / 256, 256, 0, stream>>>(wv, wvb, nW);
  cast_f32_bf16<<<(nW + 255) / 256, 256, 0, stream>>>(wo, wob, nW);
  cast_table_bf16<<<(nT + 255) / 256, 256, 0, stream>>>(rpe, tb);

  dim3 ggrid(EMB / TN, MROWS / TM);  // (8, 32)
  gemm_bf16<<<ggrid, 256, 0, stream>>>(xb, wqb, bq, qb, EMB, EMB, 0);
  gemm_bf16<<<ggrid, 256, 0, stream>>>(xb, wkb, bk, kb, EMB, EMB, 0);
  gemm_bf16<<<ggrid, 256, 0, stream>>>(xb, wvb, bv, vb, EMB, EMB, 0);

  dim3 agrid(BATCH * NH, SLEN / 64);  // (64, 16)
  attn_rpe<<<agrid, 128, 0, stream>>>(qb, kb, vb, tb, cb, 0.125f);

  gemm_bf16<<<ggrid, 256, 0, stream>>>(cb, wob, bo, d_out, EMB, EMB, 1);

  (void)in_sizes; (void)n_in; (void)out_size; (void)ws_size;
}
